// Decoder_WOG_59493886984736
// MI455X (gfx1250) — matrix operand order repaired and checked by reading, not compiled, not run on hardware
//
#include <hip/hip_runtime.h>

typedef _Float16 f16;
typedef __attribute__((ext_vector_type(8)))  _Float16 v8h;
typedef __attribute__((ext_vector_type(16))) _Float16 v16h;
typedef __attribute__((ext_vector_type(8)))  float    v8f;

#define WAVES 8
#define B_    64
#define T_    12
#define N_    2048
#define KP0   96      // layer0 weight K pad: rows [s(64) | x(2) | 0(30)]
#define KP1   128     // layer1 K: [h1(64) | s1(64)]

// LDS layout (units: halves)
#define O_WZR0 0
#define O_WH0  (O_WZR0 + 128*KP0)          // 12288
#define O_WZR1 (O_WH0  + 64*KP0)           // 18432
#define O_WH1  (O_WZR1 + 128*KP1)          // 34816
#define O_WAVE (O_WH1  + 64*KP1)           // 43008
// per-wave: insX 16x32 ([x|0] tile) + cand0 16x64 (r*s) + ins1 16x128 ([h1|s1])
#define PER_WAVE (16*32 + 16*64 + 16*128)  // 3584 halves
#define LDS_TOTAL (O_WAVE + WAVES*PER_WAVE) // 71680 halves = 140KB

#define WMMA(a,b,c) __builtin_amdgcn_wmma_f32_16x16x32_f16(false,(a),false,(b),(short)0,(c),false,false)

// A-tile (16x32 f16): m = lane&15; K(lo j) = kbase + hi*8 + j, K(hi j) = kbase + 16 + hi*8 + j
__device__ __forceinline__ v16h ldsA(const f16* ins, int Kp, int m, int kk, int hi) {
  const v8h lo = *(const v8h*)(ins + m*Kp + kk*32 + hi*8);
  const v8h hh = *(const v8h*)(ins + m*Kp + kk*32 + 16 + hi*8);
  return __builtin_shufflevector(lo, hh, 0,1,2,3,4,5,6,7,8,9,10,11,12,13,14,15);
}
// B-tile (32x16 f16): col n = lane&15; K(lo j) = kk*32 + hi*8 + j, K(hi j) = kk*32 + 16 + hi*8 + j  (weights stored transposed [n][k])
__device__ __forceinline__ v16h ldsB(const f16* wt, int Kp, int col, int kk, int hi) {
  const v8h lo = *(const v8h*)(wt + col*Kp + kk*32 + hi*8);
  const v8h hh = *(const v8h*)(wt + col*Kp + kk*32 + 16 + hi*8);
  return __builtin_shufflevector(lo, hh, 0,1,2,3,4,5,6,7,8,9,10,11,12,13,14,15);
}
// C layout: element (m = i + 8*hi, n = colbase + (lane&15)) lives in component i
__device__ __forceinline__ void storeC(f16* ins, int Kp, int colbase, int nl, int hi, v8f v) {
  #pragma unroll
  for (int i = 0; i < 8; ++i)
    ins[(i + 8*hi)*Kp + colbase + nl] = (f16)v[i];
}
__device__ __forceinline__ v8f splat8(float s) {
  v8f r;
  #pragma unroll
  for (int i = 0; i < 8; ++i) r[i] = s;
  return r;
}

extern "C" __global__ void __launch_bounds__(WAVES*32)
gru2_fused_wmma(const float* __restrict__ x,    const float* __restrict__ sinit,
                const float* __restrict__ Wzr0, const float* __restrict__ bzr0,
                const float* __restrict__ Wh0,  const float* __restrict__ bh0,
                const float* __restrict__ Wzr1, const float* __restrict__ bzr1,
                const float* __restrict__ Wh1,  const float* __restrict__ bh1,
                const float* __restrict__ Wout, const float* __restrict__ bout,
                float* __restrict__ out)
{
  __shared__ f16 lds[LDS_TOTAL];
  const int tid = threadIdx.x;

  // ---- stage weights into LDS: f16, transposed [n][k], rows reordered to [s|x], zero-padded ----
  for (int idx = tid; idx < 128*KP0; idx += WAVES*32) {           // Wzr0: (66,128) -> [128][96]
    int kk = idx / 128, n = idx % 128;
    float v = (kk < 64) ? Wzr0[(kk+2)*128 + n] : (kk < 66 ? Wzr0[(kk-64)*128 + n] : 0.f);
    lds[O_WZR0 + n*KP0 + kk] = (f16)v;
  }
  for (int idx = tid; idx < 64*KP0; idx += WAVES*32) {            // Wh0: (66,64) -> [64][96]
    int kk = idx / 64, n = idx % 64;
    float v = (kk < 64) ? Wh0[(kk+2)*64 + n] : (kk < 66 ? Wh0[(kk-64)*64 + n] : 0.f);
    lds[O_WH0 + n*KP0 + kk] = (f16)v;
  }
  for (int idx = tid; idx < 128*KP1; idx += WAVES*32) {           // Wzr1: (128,128) -> [128][128]
    int kk = idx / 128, n = idx % 128;
    lds[O_WZR1 + n*KP1 + kk] = (f16)Wzr1[kk*128 + n];
  }
  for (int idx = tid; idx < 64*KP1; idx += WAVES*32) {            // Wh1: (128,64) -> [64][128]
    int kk = idx / 64, n = idx % 64;
    lds[O_WH1 + n*KP1 + kk] = (f16)Wh1[kk*64 + n];
  }
  __syncthreads();

  const int wave = tid >> 5, lane = tid & 31;
  const int nl = lane & 15, hi = lane >> 4;
  const int tile = blockIdx.x * WAVES + wave;   // 0..8191
  const int b  = tile >> 7;                     // batch index
  const int n0 = (tile & 127) << 4;             // first of 16 sequences along N

  f16* insX  = lds + O_WAVE + wave*PER_WAVE;    // 16 x 32  : [x(2) | 0(30)]
  f16* cand0 = insX + 16*32;                    // 16 x 64  : r*s0
  f16* ins1  = cand0 + 16*64;                   // 16 x 128 : [h1/s0 | s1]

  // per-lane bias / head registers (bias depends only on n -> splat into C init)
  float bzr0_r[8], bzr1_r[8], bh0_r[4], bh1_r[4], wout_r[4];
  #pragma unroll
  for (int j = 0; j < 8; ++j) { bzr0_r[j] = bzr0[j*16 + nl]; bzr1_r[j] = bzr1[j*16 + nl]; }
  #pragma unroll
  for (int j = 0; j < 4; ++j) { bh0_r[j] = bh0[j*16 + nl]; bh1_r[j] = bh1[j*16 + nl]; wout_r[j] = Wout[j*16 + nl]; }
  const float bo = bout[0];

  // ---- init recurrent state: f32 master in C layout + f16 copy in ins1 = [s0 | s1] ----
  v8f s0c[4], s1c[4];
  #pragma unroll
  for (int j = 0; j < 4; ++j) {
    #pragma unroll
    for (int i = 0; i < 8; ++i) {
      int m = i + 8*hi;
      float v0 = sinit[(((size_t)0*B_ + b)*N_ + n0 + m)*64 + j*16 + nl];
      float v1 = sinit[(((size_t)1*B_ + b)*N_ + n0 + m)*64 + j*16 + nl];
      s0c[j][i] = v0; s1c[j][i] = v1;
      ins1[m*KP1 + j*16 + nl]      = (f16)v0;
      ins1[m*KP1 + 64 + j*16 + nl] = (f16)v1;
    }
  }
  if (lane < 16) {  // zero pad cols 2..31 of insX (written once, never touched again)
    for (int c = 2; c < 32; ++c) insX[lane*32 + c] = (f16)0.f;
  }

  for (int t = 0; t < T_; ++t) {
    // stage x_t into insX cols 0..1 (one row per lane)
    if (lane < 16) {
      const float* xp = x + ((((size_t)b*T_ + t)*N_) + n0 + lane)*2;
      insX[lane*32 + 0] = (f16)xp[0];
      insX[lane*32 + 1] = (f16)xp[1];
    }

    // ---------- layer 0: z_r = [s0|x] @ Wzr0 + bzr0 ----------
    v16h aS0 = ldsA(ins1, KP1, nl, 0, hi);      // s0 cols  0..31
    v16h aS1 = ldsA(ins1, KP1, nl, 1, hi);      // s0 cols 32..63
    v16h aX  = ldsA(insX, 32,  nl, 0, hi);      // [x|0] tile (weight rows 64..95)
    v8f zc[4];
    #pragma unroll
    for (int j = 0; j < 8; ++j) {
      v8f acc = splat8(bzr0_r[j]);
      acc = WMMA(aS0, ldsB(lds + O_WZR0, KP0, j*16 + nl, 0, hi), acc);
      acc = WMMA(aS1, ldsB(lds + O_WZR0, KP0, j*16 + nl, 1, hi), acc);
      acc = WMMA(aX,  ldsB(lds + O_WZR0, KP0, j*16 + nl, 2, hi), acc);
      if (j < 4) zc[j] = acc;
      else       storeC(cand0, 64, (j-4)*16, nl, hi, acc * s0c[j-4]);  // r*s0
    }
    // cand = [r*s0 | x] @ Wh0 + bh0 ; layer-0 state update
    v16h aC0 = ldsA(cand0, 64, nl, 0, hi);
    v16h aC1 = ldsA(cand0, 64, nl, 1, hi);
    #pragma unroll
    for (int j = 0; j < 4; ++j) {
      v8f acc = splat8(bh0_r[j]);
      acc = WMMA(aC0, ldsB(lds + O_WH0, KP0, j*16 + nl, 0, hi), acc);
      acc = WMMA(aC1, ldsB(lds + O_WH0, KP0, j*16 + nl, 1, hi), acc);
      acc = WMMA(aX,  ldsB(lds + O_WH0, KP0, j*16 + nl, 2, hi), acc);
      s0c[j] = zc[j]*s0c[j] + (splat8(1.f) - zc[j])*acc;
      storeC(ins1, KP1, j*16, nl, hi, s0c[j]);   // h1_t == next-step s0 (single store)
    }

    // ---------- layer 1: z_r = [h1|s1] @ Wzr1 + bzr1 ----------
    v16h a1[4];
    #pragma unroll
    for (int kk = 0; kk < 4; ++kk) a1[kk] = ldsA(ins1, KP1, nl, kk, hi);
    v8f z1[4];
    #pragma unroll
    for (int j = 0; j < 8; ++j) {
      v8f acc = splat8(bzr1_r[j]);
      #pragma unroll
      for (int kk = 0; kk < 4; ++kk)
        acc = WMMA(a1[kk], ldsB(lds + O_WZR1, KP1, j*16 + nl, kk, hi), acc);
      if (j < 4) z1[j] = acc;
      else       storeC(ins1, KP1, 64 + (j-4)*16, nl, hi, acc * s1c[j-4]);  // r1*s1
    }
    a1[2] = ldsA(ins1, KP1, nl, 2, hi);
    a1[3] = ldsA(ins1, KP1, nl, 3, hi);
    #pragma unroll
    for (int j = 0; j < 4; ++j) {
      v8f acc = splat8(bh1_r[j]);
      #pragma unroll
      for (int kk = 0; kk < 4; ++kk)
        acc = WMMA(a1[kk], ldsB(lds + O_WH1, KP1, j*16 + nl, kk, hi), acc);
      s1c[j] = z1[j]*s1c[j] + (splat8(1.f) - z1[j])*acc;
      storeC(ins1, KP1, 64 + j*16, nl, hi, s1c[j]);  // next-step s1
    }

    // ---------- output head: out = s1 @ Wout + bout ----------
    float p[8];
    #pragma unroll
    for (int i = 0; i < 8; ++i)
      p[i] = s1c[0][i]*wout_r[0] + s1c[1][i]*wout_r[1] + s1c[2][i]*wout_r[2] + s1c[3][i]*wout_r[3];
    #pragma unroll
    for (int mask = 1; mask < 16; mask <<= 1) {
      #pragma unroll
      for (int i = 0; i < 8; ++i) p[i] += __shfl_xor(p[i], mask, 32);
    }
    if (nl == 0) {
      float* op = out + (((size_t)b*T_ + t)*N_) + n0 + 8*hi;
      #pragma unroll
      for (int i = 0; i < 8; ++i) op[i] = p[i] + bo;
    }
  }
}

extern "C" void kernel_launch(void* const* d_in, const int* in_sizes, int n_in,
                              void* d_out, int out_size, void* d_ws, size_t ws_size,
                              hipStream_t stream) {
  (void)in_sizes; (void)n_in; (void)d_ws; (void)ws_size; (void)out_size;
  const float* x    = (const float*)d_in[0];
  const float* si   = (const float*)d_in[1];
  const float* Wzr0 = (const float*)d_in[2];
  const float* bzr0 = (const float*)d_in[3];
  const float* Wh0  = (const float*)d_in[4];
  const float* bh0  = (const float*)d_in[5];
  const float* Wzr1 = (const float*)d_in[6];
  const float* bzr1 = (const float*)d_in[7];
  const float* Wh1  = (const float*)d_in[8];
  const float* bh1  = (const float*)d_in[9];
  const float* Wout = (const float*)d_in[10];
  const float* bout = (const float*)d_in[11];
  float* out = (float*)d_out;

  const int tiles = B_ * (N_ / 16);           // 8192 sequence tiles (16 rows each)
  const int blocks = tiles / WAVES;           // 1024 workgroups of 8 waves
  hipLaunchKernelGGL(gru2_fused_wmma, dim3(blocks), dim3(WAVES*32), 0, stream,
                     x, si, Wzr0, bzr0, Wh0, bh0, Wzr1, bzr1, Wh1, bh1, Wout, bout, out);
}
